// ChebGCN3Multi_fusion_63024350101698
// MI455X (gfx1250) — compile-verified, matrix operand order checked by reading, not run on hardware
//
#include <hip/hip_runtime.h>
#include <math.h>

// ---------------------------------------------------------------------------
// ChebGCN3Multi on MI455X (gfx1250, wave32, WMMA).
// Strategy: densify Lhat (4096x4096, L2-resident, converted once to bf16),
// run EVERYTHING as bf16 WMMA GEMMs (v_wmma_f32_16x16x32_bf16), fuse the
// Chebyshev recurrence and GraphNorm into GEMM epilogues / per-column passes.
// The Lhat GEMMs (constant bf16 A) stage their A tiles with
// global_load_async_to_lds_b128 (ASYNCcnt path) — no VGPR round-trip.
// ~3.6 TFLOP of matrix work, compute-bound on the matrix pipes.
// ---------------------------------------------------------------------------

#define NNODES 4096
#define NEDGES 65536
#define GN_EPS 1e-5f

typedef __attribute__((ext_vector_type(16))) __bf16 v16bf;
typedef __attribute__((ext_vector_type(8)))  float  v8f;

union FragU { v16bf v; unsigned int u[8]; };

__device__ __forceinline__ unsigned short f2bf(float f) {
    unsigned int u = __float_as_uint(f);
    unsigned int r = u + 0x7FFFu + ((u >> 16) & 1u);   // round-to-nearest-even
    return (unsigned short)(r >> 16);
}

// ---------------------------------------------------------------------------
// Shared epilogue: D = alpha*acc + beta*C with the documented 16x16 f32 C/D
// layout (VGPR r -> M = r + (lane>=16)*8, N = lane%16).
// ---------------------------------------------------------------------------
__device__ __forceinline__ void wmma_epilogue(
    v8f acc[2][4], const float* __restrict__ C, float* __restrict__ D,
    int Nc, int rowBase, int colBase, int wm, int wn, int half, int lr,
    float alpha, float beta)
{
    #pragma unroll
    for (int mt = 0; mt < 2; ++mt) {
        #pragma unroll
        for (int nt = 0; nt < 4; ++nt) {
            #pragma unroll
            for (int r = 0; r < 8; ++r) {
                int grow = rowBase + wm * 32 + mt * 16 + half * 8 + r;
                int gcol = colBase + wn * 64 + nt * 16 + lr;
                size_t o = (size_t)grow * Nc + gcol;
                float v = alpha * acc[mt][nt][r];
                if (beta != 0.0f) v += beta * C[o];
                D[o] = v;
            }
        }
    }
}

// ---------------------------------------------------------------------------
// Generic bf16-WMMA GEMM:  D[M,Nc] = alpha * (A[M,K] @ B[K,Nc]) + beta * C
// A,B,C,D f32 row-major; A/B converted to bf16 while staging to LDS.
// Requires M%128==0, Nc%128==0, K%32==0 (true for all calls).
// Block = 256 threads = 8 waves; tile 128x128, BK=32; wave grid 4(M) x 2(N),
// each wave owns 32x64 via 2x4 WMMA 16x16 accumulators.
// ---------------------------------------------------------------------------
__global__ __launch_bounds__(256)
void gemm_bf16_wmma(const float* __restrict__ A, const float* __restrict__ B,
                    const float* __restrict__ C, float* __restrict__ D,
                    int M, int Nc, int Kd, float alpha, float beta)
{
    __shared__ __align__(16) unsigned short As[128 * 32];  // [row][k]
    __shared__ __align__(16) unsigned short Bs[128 * 32];  // [col][k] (B transposed)

    const int tid  = threadIdx.x;
    const int wid  = tid >> 5;
    const int lane = tid & 31;
    const int half = lane >> 4;     // K-half selector per 16-bit A/B layout
    const int lr   = lane & 15;

    const int wm = wid & 3;
    const int wn = wid >> 2;
    const int rowBase = blockIdx.y * 128;
    const int colBase = blockIdx.x * 128;

    v8f acc[2][4];
    for (int mt = 0; mt < 2; ++mt)
        for (int nt = 0; nt < 4; ++nt)
            acc[mt][nt] = (v8f)(0.0f);

    for (int k0 = 0; k0 < Kd; k0 += 32) {
        // ---- issue all tile loads first (batched under one loadcnt wait) ----
        float4 at[4], bt[4];
        #pragma unroll
        for (int i = 0; i < 4; ++i) {
            int idx = i * 256 + tid;               // A: 128x32, 1024 float4
            at[i] = *(const float4*)(A + (size_t)(rowBase + (idx >> 3)) * Kd +
                                     k0 + ((idx & 7) << 2));
        }
        #pragma unroll
        for (int i = 0; i < 4; ++i) {
            int idx = i * 256 + tid;               // B: 32x128, 1024 float4
            bt[i] = *(const float4*)(B + (size_t)(k0 + (idx >> 5)) * Nc +
                                     colBase + ((idx & 31) << 2));
        }
        if (k0 + 32 < Kd)   // prefetch next A K-tile (global_prefetch_b8)
            __builtin_prefetch(A + (size_t)(rowBase + (tid >> 1)) * Kd + k0 + 32, 0, 0);

        // ---- convert + store to LDS ----
        #pragma unroll
        for (int i = 0; i < 4; ++i) {
            int idx = i * 256 + tid;
            unsigned short* dst = &As[(idx >> 3) * 32 + ((idx & 7) << 2)];
            dst[0] = f2bf(at[i].x); dst[1] = f2bf(at[i].y);
            dst[2] = f2bf(at[i].z); dst[3] = f2bf(at[i].w);
        }
        #pragma unroll
        for (int i = 0; i < 4; ++i) {
            int idx = i * 256 + tid;
            int kr = idx >> 5;
            int nc = (idx & 31) << 2;
            Bs[(nc + 0) * 32 + kr] = f2bf(bt[i].x);
            Bs[(nc + 1) * 32 + kr] = f2bf(bt[i].y);
            Bs[(nc + 2) * 32 + kr] = f2bf(bt[i].z);
            Bs[(nc + 3) * 32 + kr] = f2bf(bt[i].w);
        }
        __syncthreads();

        // ---- fragments per documented 16-bit layout; 8 WMMAs per K-step ----
        FragU af[2], bf[4];
        #pragma unroll
        for (int mt = 0; mt < 2; ++mt) {
            int arow = wm * 32 + mt * 16 + lr;
            #pragma unroll
            for (int vg = 0; vg < 8; ++vg) {
                int kb = ((vg & 4) << 2) + half * 8 + ((vg & 3) << 1);
                af[mt].u[vg] = *(const unsigned int*)&As[arow * 32 + kb];
            }
        }
        #pragma unroll
        for (int nt = 0; nt < 4; ++nt) {
            int bcol = wn * 64 + nt * 16 + lr;
            #pragma unroll
            for (int vg = 0; vg < 8; ++vg) {
                int kb = ((vg & 4) << 2) + half * 8 + ((vg & 3) << 1);
                bf[nt].u[vg] = *(const unsigned int*)&Bs[bcol * 32 + kb];
            }
        }
        #pragma unroll
        for (int mt = 0; mt < 2; ++mt)
            #pragma unroll
            for (int nt = 0; nt < 4; ++nt)
                acc[mt][nt] = __builtin_amdgcn_wmma_f32_16x16x32_bf16(
                    false, af[mt].v, false, bf[nt].v,
                    (short)0, acc[mt][nt], false, false);
        __syncthreads();
    }

    wmma_epilogue(acc, C, D, Nc, rowBase, colBase, wm, wn, half, lr, alpha, beta);
}

// ---------------------------------------------------------------------------
// Lhat-GEMM variant: A is already bf16 (persistent Lhat_bf16). A tiles are
// staged straight into LDS with global_load_async_to_lds_b128 (ASYNCcnt),
// skipping the VGPR round-trip and any A-side conversion VALU work.
// D = alpha * (Abf @ B) + beta * C;  B/C/D are f32.
// ---------------------------------------------------------------------------
__global__ __launch_bounds__(256)
void gemm_abf16_async_wmma(const unsigned short* __restrict__ Abf,
                           const float* __restrict__ B,
                           const float* __restrict__ C, float* __restrict__ D,
                           int M, int Nc, int Kd, float alpha, float beta)
{
    __shared__ __align__(16) unsigned short As[128 * 32];  // [row][k]
    __shared__ __align__(16) unsigned short Bs[128 * 32];  // [col][k]

    const int tid  = threadIdx.x;
    const int wid  = tid >> 5;
    const int lane = tid & 31;
    const int half = lane >> 4;
    const int lr   = lane & 15;

    const int wm = wid & 3;
    const int wn = wid >> 2;
    const int rowBase = blockIdx.y * 128;
    const int colBase = blockIdx.x * 128;

    // Generic LDS pointer low 32 bits == LDS byte offset (aperture rule).
    const unsigned ldsA = (unsigned)(size_t)(&As[0]);

    v8f acc[2][4];
    for (int mt = 0; mt < 2; ++mt)
        for (int nt = 0; nt < 4; ++nt)
            acc[mt][nt] = (v8f)(0.0f);

    for (int k0 = 0; k0 < Kd; k0 += 32) {
        // ---- A tile (128 x 32 bf16 = 8 KB): 512 16B chunks, 2 per thread,
        //      async copy global -> LDS (no VGPR data) ----
        #pragma unroll
        for (int i = 0; i < 2; ++i) {
            int idx = i * 256 + tid;
            int r  = idx >> 2;               // 4 chunks (16B = 8 bf16) per row
            int ce = (idx & 3) << 3;         // chunk start element in row
            unsigned dst = ldsA + (unsigned)((r * 32 + ce) * 2);
            unsigned long long ga =
                (unsigned long long)(Abf + (size_t)(rowBase + r) * Kd + k0 + ce);
            asm volatile("global_load_async_to_lds_b128 %0, %1, off"
                         :: "v"(dst), "v"(ga) : "memory");
        }

        // ---- B tile f32 -> bf16 (transposed into Bs[col][k]) ----
        float4 bt[4];
        #pragma unroll
        for (int i = 0; i < 4; ++i) {
            int idx = i * 256 + tid;
            bt[i] = *(const float4*)(B + (size_t)(k0 + (idx >> 5)) * Nc +
                                     colBase + ((idx & 31) << 2));
        }
        if (k0 + 32 < Kd)
            __builtin_prefetch(B + (size_t)(k0 + 32 + (tid >> 3)) * Nc + colBase, 0, 0);
        #pragma unroll
        for (int i = 0; i < 4; ++i) {
            int idx = i * 256 + tid;
            int kr = idx >> 5;
            int nc = (idx & 31) << 2;
            Bs[(nc + 0) * 32 + kr] = f2bf(bt[i].x);
            Bs[(nc + 1) * 32 + kr] = f2bf(bt[i].y);
            Bs[(nc + 2) * 32 + kr] = f2bf(bt[i].z);
            Bs[(nc + 3) * 32 + kr] = f2bf(bt[i].w);
        }

        asm volatile("s_wait_asynccnt 0x0" ::: "memory");  // A tile landed in LDS
        __syncthreads();

        FragU af[2], bf[4];
        #pragma unroll
        for (int mt = 0; mt < 2; ++mt) {
            int arow = wm * 32 + mt * 16 + lr;
            #pragma unroll
            for (int vg = 0; vg < 8; ++vg) {
                int kb = ((vg & 4) << 2) + half * 8 + ((vg & 3) << 1);
                af[mt].u[vg] = *(const unsigned int*)&As[arow * 32 + kb];
            }
        }
        #pragma unroll
        for (int nt = 0; nt < 4; ++nt) {
            int bcol = wn * 64 + nt * 16 + lr;
            #pragma unroll
            for (int vg = 0; vg < 8; ++vg) {
                int kb = ((vg & 4) << 2) + half * 8 + ((vg & 3) << 1);
                bf[nt].u[vg] = *(const unsigned int*)&Bs[bcol * 32 + kb];
            }
        }
        #pragma unroll
        for (int mt = 0; mt < 2; ++mt)
            #pragma unroll
            for (int nt = 0; nt < 4; ++nt)
                acc[mt][nt] = __builtin_amdgcn_wmma_f32_16x16x32_bf16(
                    false, af[mt].v, false, bf[nt].v,
                    (short)0, acc[mt][nt], false, false);
        __syncthreads();
    }

    wmma_epilogue(acc, C, D, Nc, rowBase, colBase, wm, wn, half, lr, alpha, beta);
}

// ---------------------------------------------------------------------------
// Small helper kernels
// ---------------------------------------------------------------------------
__global__ void zero_kernel(float* p, size_t n) {
    size_t i = (size_t)blockIdx.x * blockDim.x + threadIdx.x;
    if (i < n) p[i] = 0.0f;
}

__global__ void cvt_bf16_kernel(const float* __restrict__ src,
                                unsigned short* __restrict__ dst, size_t n) {
    size_t i = (size_t)blockIdx.x * blockDim.x + threadIdx.x;
    if (i < n) dst[i] = f2bf(src[i]);
}

__global__ void deg_kernel(const int* __restrict__ row, float* deg, int E) {
    int i = blockIdx.x * blockDim.x + threadIdx.x;
    if (i < E) atomicAdd(&deg[row[i]], 1.0f);
}

__global__ void dinv_kernel(float* deg, int N) {
    int i = blockIdx.x * blockDim.x + threadIdx.x;
    if (i < N) {
        float d = deg[i];
        deg[i] = (d > 0.0f) ? rsqrtf(fmaxf(d, 1.0f)) : 0.0f;
    }
}

__global__ void lhat_scatter(const int* __restrict__ row, const int* __restrict__ col,
                             const float* __restrict__ dinv, float* Lhat, int E, int N) {
    int i = blockIdx.x * blockDim.x + threadIdx.x;
    if (i < E) {
        int r = row[i], c = col[i];
        atomicAdd(&Lhat[(size_t)r * N + c], -dinv[r] * dinv[c]);
    }
}

__global__ void colstats(const float* __restrict__ X, int M, int F,
                         float* __restrict__ s, float* __restrict__ ss) {
    int c = blockIdx.x * blockDim.x + threadIdx.x;
    if (c >= F) return;
    float a = 0.0f, b = 0.0f;
    for (int r = 0; r < M; ++r) {
        float v = X[(size_t)r * F + c];
        a += v; b += v * v;
    }
    s[c] = a; ss[c] = b;
}

// Fold conv bias cb into GraphNorm analytically: x' = acc + cb.
// y = gw * (x' - mean'*gm) * rsqrt(var + eps) + gb  ==  colA*acc + colC
__global__ void gn_finalize(const float* __restrict__ s, const float* __restrict__ ss,
                            const float* __restrict__ cb, const float* __restrict__ gw,
                            const float* __restrict__ gb, const float* __restrict__ gm,
                            int M, int F, float* __restrict__ colA, float* __restrict__ colC) {
    int c = blockIdx.x * blockDim.x + threadIdx.x;
    if (c >= F) return;
    float invM  = 1.0f / (float)M;
    float m0    = s[c] * invM;
    float mean  = m0 + cb[c];
    float ex2   = ss[c] * invM + 2.0f * cb[c] * m0 + cb[c] * cb[c];
    float sub   = mean * gm[c];
    float var   = ex2 - 2.0f * sub * mean + sub * sub;
    float scale = gw[c] * rsqrtf(var + GN_EPS);
    colA[c] = scale;
    colC[c] = scale * (cb[c] - sub) + gb[c];
}

// act==0: leaky_relu(y, 0.1); act==1: relu(resid + y)
__global__ void gn_apply(const float* __restrict__ ACC, const float* __restrict__ colA,
                         const float* __restrict__ colC, const float* __restrict__ resid,
                         float* __restrict__ X, size_t total, int F, int act) {
    size_t i = (size_t)blockIdx.x * blockDim.x + threadIdx.x;
    if (i >= total) return;
    int c = (int)(i % F);
    float y = colA[c] * ACC[i] + colC[c];
    if (act == 0) {
        y = (y > 0.0f) ? y : 0.1f * y;
    } else {
        y += resid[i];
        y = fmaxf(y, 0.0f);
    }
    X[i] = y;
}

__global__ void colmean(const float* __restrict__ X, int M, int F, float* __restrict__ g) {
    int c = blockIdx.x * blockDim.x + threadIdx.x;
    if (c >= F) return;
    float a = 0.0f;
    for (int r = 0; r < M; ++r) a += X[(size_t)r * F + c];
    g[c] = a / (float)M;
}

__global__ void head_kernel(const float* __restrict__ g, const float* __restrict__ lw,
                            const float* __restrict__ lb, float* __restrict__ out,
                            int do_softplus) {
    __shared__ float red[256];
    __shared__ float ov[10];
    int tid = threadIdx.x;
    for (int j = 0; j < 10; ++j) {
        float p = 0.0f;
        for (int i = tid; i < 4096; i += 256)
            p += fmaxf(g[i], 0.0f) * lw[i * 10 + j];
        red[tid] = p;
        __syncthreads();
        for (int s2 = 128; s2 > 0; s2 >>= 1) {
            if (tid < s2) red[tid] += red[tid + s2];
            __syncthreads();
        }
        if (tid == 0) ov[j] = red[0] + lb[j];
        __syncthreads();
    }
    if (tid == 0) {
        float o[10];
        float mx = -1e30f;
        for (int j = 0; j < 10; ++j) {
            float v = ov[j];
            if (do_softplus) v = (v > 20.0f) ? v : log1pf(expf(v));
            o[j] = v;
            mx = fmaxf(mx, v);
        }
        float se = 0.0f, e[10];
        for (int j = 0; j < 10; ++j) { e[j] = expf(o[j] - mx); se += e[j]; }
        float inv = 1.0f / se;
        for (int j = 0; j < 10; ++j)
            out[j] = e[j] * inv * fmaxf(o[j], 0.0f);
    }
}

// ---------------------------------------------------------------------------
// Host-side orchestration
// ---------------------------------------------------------------------------
static inline void launch_gemm(hipStream_t s, const float* A, const float* B,
                               const float* C, float* D,
                               int M, int Nc, int K, float alpha, float beta) {
    dim3 grid(Nc / 128, M / 128);
    gemm_bf16_wmma<<<grid, 256, 0, s>>>(A, B, C, D, M, Nc, K, alpha, beta);
}

static inline void launch_gemm_lhat(hipStream_t s, const unsigned short* Abf,
                                    const float* B, const float* C, float* D,
                                    int M, int Nc, int K, float alpha, float beta) {
    dim3 grid(Nc / 128, M / 128);
    gemm_abf16_async_wmma<<<grid, 256, 0, s>>>(Abf, B, C, D, M, Nc, K, alpha, beta);
}

extern "C" void kernel_launch(void* const* d_in, const int* in_sizes, int n_in,
                              void* d_out, int out_size, void* d_ws, size_t ws_size,
                              hipStream_t stream) {
    const int N = NNODES, E = NEDGES;
    const int dims[5] = {4096, 2048, 1024, 512, 4096};

    const int*   edge = (const int*)d_in[0];
    const int*   erow = edge;
    const int*   ecol = edge + E;
    const float* feats[3] = { (const float*)d_in[1], (const float*)d_in[2],
                              (const float*)d_in[3] };

    // ---- workspace layout ----
    float* ws = (float*)d_ws;
    const size_t NN = (size_t)N * 4096;
    float* Lhat = ws;                // N x N f32 (atomic build target)
    float* X    = ws + 1 * NN;
    float* T1   = ws + 2 * NN;
    float* T2   = ws + 3 * NN;
    float* ACC  = ws + 4 * NN;
    float* deg  = ws + 5 * NN;       // N (becomes dinv)
    float* csum = deg  + N;
    float* csq  = csum + 4096;
    float* colA = csq  + 4096;
    float* colC = colA + 4096;
    float* gbuf = colC + 4096;
    unsigned short* Lhat_bf = (unsigned short*)(gbuf + 4096);  // N x N bf16
    float* dout = (float*)d_out;

    // ---- build dense scaled-Laplacian Lhat, convert once to bf16 ----
    {
        size_t nL = (size_t)N * N;
        zero_kernel<<<(unsigned)((nL + 255) / 256), 256, 0, stream>>>(Lhat, nL);
        zero_kernel<<<(N + 255) / 256, 256, 0, stream>>>(deg, (size_t)N);
        deg_kernel<<<(E + 255) / 256, 256, 0, stream>>>(erow, deg, E);
        dinv_kernel<<<(N + 255) / 256, 256, 0, stream>>>(deg, N);
        lhat_scatter<<<(E + 255) / 256, 256, 0, stream>>>(erow, ecol, deg, Lhat, E, N);
        cvt_bf16_kernel<<<(unsigned)((nL + 255) / 256), 256, 0, stream>>>(Lhat, Lhat_bf, nL);
    }

    for (int b = 0; b < 3; ++b) {
        const int base = 4 + b * 22;
        const float* Wk[4]; const float* cbp[4]; const float* gwp[4];
        const float* gbp[4]; const float* gmp[4];
        for (int k = 0; k < 4; ++k) {
            Wk[k]  = (const float*)d_in[base + k];        // [4, fin, fout]
            cbp[k] = (const float*)d_in[base + 4 + k];
            gwp[k] = (const float*)d_in[base + 8 + k];
            gbp[k] = (const float*)d_in[base + 12 + k];
            gmp[k] = (const float*)d_in[base + 16 + k];
        }
        const float* lw = (const float*)d_in[base + 20];
        const float* lb = (const float*)d_in[base + 21];

        hipMemcpyAsync(X, feats[b], NN * sizeof(float),
                       hipMemcpyDeviceToDevice, stream);

        for (int i = 0; i < 4; ++i) {
            const int fin = dims[i], fout = dims[i + 1];
            const size_t wsz = (size_t)fin * fout;
            const float* W = Wk[i];

            // ACC = T0 @ W0          (T0 == X)
            launch_gemm(stream, X, W, nullptr, ACC, N, fout, fin, 1.0f, 0.0f);
            // T1 = Lhat @ X                        (async bf16-A path)
            launch_gemm_lhat(stream, Lhat_bf, X, nullptr, T1, N, fin, N, 1.0f, 0.0f);
            // ACC += T1 @ W1
            launch_gemm(stream, T1, W + wsz, ACC, ACC, N, fout, fin, 1.0f, 1.0f);
            // T2 = 2*Lhat@T1 - T0                  (async bf16-A path)
            launch_gemm_lhat(stream, Lhat_bf, T1, X, T2, N, fin, N, 2.0f, -1.0f);
            // ACC += T2 @ W2
            launch_gemm(stream, T2, W + 2 * wsz, ACC, ACC, N, fout, fin, 1.0f, 1.0f);
            // T3 = 2*Lhat@T2 - T1   (into T1; C==D read-before-write is safe)
            launch_gemm_lhat(stream, Lhat_bf, T2, T1, T1, N, fin, N, 2.0f, -1.0f);
            // ACC += T3 @ W3
            launch_gemm(stream, T1, W + 3 * wsz, ACC, ACC, N, fout, fin, 1.0f, 1.0f);

            // bias + GraphNorm + activation (+residual on layer 4)
            colstats<<<(fout + 255) / 256, 256, 0, stream>>>(ACC, N, fout, csum, csq);
            gn_finalize<<<(fout + 255) / 256, 256, 0, stream>>>(
                csum, csq, cbp[i], gwp[i], gbp[i], gmp[i], N, fout, colA, colC);
            size_t total = (size_t)N * fout;
            gn_apply<<<(unsigned)((total + 255) / 256), 256, 0, stream>>>(
                ACC, colA, colC, (i == 3) ? feats[b] : nullptr, X, total, fout,
                (i == 3) ? 1 : 0);
        }

        // global mean pool + linear head + softmax*relu
        colmean<<<(4096 + 255) / 256, 256, 0, stream>>>(X, N, 4096, gbuf);
        head_kernel<<<1, 256, 0, stream>>>(gbuf, lw, lb, dout + b * 10,
                                           (b == 0) ? 1 : 0);
    }
    (void)in_sizes; (void)n_in; (void)out_size; (void)ws_size;
}
